// ODEFunc_A_30906584662177
// MI455X (gfx1250) — matrix operand order repaired and checked by reading, not compiled, not run on hardware
//
#include <hip/hip_runtime.h>
#include <stdint.h>
#include <stddef.h>

// ---- CDNA5 WMMA types ----
typedef __attribute__((ext_vector_type(16))) __bf16  v16bf;
typedef __attribute__((ext_vector_type(8)))  float   v8f;
typedef __attribute__((ext_vector_type(4)))  float   f32x4;
typedef __attribute__((ext_vector_type(4)))  unsigned int u32x4;

#define NDIM 8192
#define HDIM 128
#define WAVES 4          // 4 waves/block, 16 rows each -> 64 rows/block -> 128 blocks

// CDNA5 async global->LDS path (ASYNCcnt), with safe fallback to sync copy.
#if defined(__AMDGCN__) && __has_builtin(__builtin_amdgcn_global_load_async_to_lds_b128)
#define USE_ASYNC_LDS 1
#else
#define USE_ASYNC_LDS 0
#endif

union BFrag {
    v16bf v;
    unsigned short h[16];
    u32x4 q[2];
};

struct AChunk { f32x4 v0, v1, v2, v3; };

__device__ __forceinline__ unsigned short f2bf(float f) {
    union { float f; uint32_t u; } c; c.f = f;
    uint32_t u = c.u;
    u += 0x7fffu + ((u >> 16) & 1u);   // round-to-nearest-even
    return (unsigned short)(u >> 16);
}

// Stage one 8 KB x-slice (2 K-blocks x 8 col tiles) into LDS.
// Async path: 4 x global_load_async_to_lds_b128 per wave (ASYNCcnt),
// sync fallback: b128 loads + ds_store_b128.
#if USE_ASYNC_LDS
typedef int v4i_vs __attribute__((vector_size(16)));   // builtin's b128 payload type
__device__ __forceinline__ void stage_tile(const unsigned short* g,
                                           unsigned short* l, int tid) {
    __attribute__((address_space(1))) v4i_vs* gp =
        (__attribute__((address_space(1))) v4i_vs*)g + tid;
    __attribute__((address_space(3))) v4i_vs* lp =
        (__attribute__((address_space(3))) v4i_vs*)l + tid;
    __builtin_amdgcn_global_load_async_to_lds_b128(gp, lp, 0, 0);
    __builtin_amdgcn_global_load_async_to_lds_b128(gp, lp, 2048, 0);
    __builtin_amdgcn_global_load_async_to_lds_b128(gp, lp, 4096, 0);
    __builtin_amdgcn_global_load_async_to_lds_b128(gp, lp, 6144, 0);
}
#else
__device__ __forceinline__ void stage_tile(const unsigned short* g,
                                           unsigned short* l, int tid) {
    const u32x4* gs = (const u32x4*)g + tid;
    u32x4* ld = (u32x4*)l + tid;
    ld[0]   = gs[0];
    ld[128] = gs[128];
    ld[256] = gs[256];
    ld[384] = gs[384];
}
#endif

__device__ __forceinline__ void wait_stage_keep4() {
#if USE_ASYNC_LDS
#if __has_builtin(__builtin_amdgcn_s_wait_asynccnt)
    __builtin_amdgcn_s_wait_asynccnt(4);   // current tile done, next may be in flight
#else
    asm volatile("s_wait_asynccnt 4" ::: "memory");
#endif
#endif
}
__device__ __forceinline__ void wait_stage_zero() {
#if USE_ASYNC_LDS
#if __has_builtin(__builtin_amdgcn_s_wait_asynccnt)
    __builtin_amdgcn_s_wait_asynccnt(0);
#else
    asm volatile("s_wait_asynccnt 0" ::: "memory");
#endif
#endif
}

__device__ __forceinline__ AChunk load_a(const float* Arow, int k, int koff1, int koff2) {
    AChunk c;
    const f32x4* p1 = (const f32x4*)(Arow + k + koff1);
    const f32x4* p2 = (const f32x4*)(Arow + k + koff2);
    c.v0 = p1[0]; c.v1 = p1[1]; c.v2 = p2[0]; c.v3 = p2[1];
    return c;
}

// ---------------------------------------------------------------------------
// Prep: convert x (N x H fp32) and W (H x H fp32) into WMMA-B-fragment tiled
// bf16 layout:  chunk[kb][ct][n][e] = src[kb*16+e][ct*16+n]  (B = x, K x N)
//               wchunk[kb][ct][n][e] = W[ct*16+n][kb*16+e]   (B = W^T)
// Each chunk is 16 contiguous bf16 (32 B) = exactly one lane's B-frag half.
// ---------------------------------------------------------------------------
__global__ __launch_bounds__(256) void prep_kernel(
    const float* __restrict__ x, const float* __restrict__ W,
    unsigned short* __restrict__ xw, unsigned short* __restrict__ ww)
{
    const int XCHUNKS = (NDIM / 16) * (HDIM / 16) * 16;   // 65536
    const int WCHUNKS = (HDIM / 16) * (HDIM / 16) * 16;   // 1024
    int tid = blockIdx.x * blockDim.x + threadIdx.x;
    if (tid < XCHUNKS) {
        int n  = tid & 15;
        int ct = (tid >> 4) & 7;
        int kb = tid >> 7;
        unsigned short* dst = xw + (size_t)tid * 16;
        const float* src = x + (size_t)(kb * 16) * HDIM + ct * 16 + n;
        #pragma unroll
        for (int e = 0; e < 16; ++e) dst[e] = f2bf(src[(size_t)e * HDIM]);
    } else if (tid < XCHUNKS + WCHUNKS) {
        int t  = tid - XCHUNKS;
        int n  = t & 15;
        int ct = (t >> 4) & 7;
        int kb = t >> 7;
        unsigned short* dst = ww + (size_t)t * 16;
        const float* src = W + (size_t)(ct * 16 + n) * HDIM + kb * 16;
        #pragma unroll
        for (int e = 0; e < 16; ++e) dst[e] = f2bf(src[e]);
    }
}

// ---------------------------------------------------------------------------
// Fused: y = A@x (bf16 WMMA, fp32 acc) + exact fp32 rowsum(A) in same pass,
// then x1 = (alpha*y + (1-alpha)*x) / (alpha*rowsum + (1-alpha)),
// then out = relu(x1 @ W^T + b) with a second WMMA stage.
// Double-buffered LDS x-slices + async global->LDS + pipelined A loads.
// ---------------------------------------------------------------------------
__global__ __launch_bounds__(32 * WAVES) void odefunc_kernel(
    const float* __restrict__ A, const float* __restrict__ x,
    const unsigned short* __restrict__ xw, const unsigned short* __restrict__ ww,
    const float* __restrict__ alpha_p, const float* __restrict__ bias,
    float* __restrict__ out)
{
    __shared__ __align__(16) unsigned short ldsX[2][2 * 8 * 16 * 16]; // 2 x 8 KB
    __shared__ __align__(16) float ldsY[WAVES][16][HDIM];             // 32 KB x1 staging

    const int tid  = threadIdx.x;
    const int wave = tid >> 5;
    const int lane = tid & 31;
    const int ln   = lane & 15;
    const int hi   = lane >> 4;            // 0: K{0-15}, 1: K{16-31}

    const int rowBase = blockIdx.x * (16 * WAVES) + wave * 16;
    const int arow    = rowBase + ln;      // A-fragment row for this lane
    const float alpha = alpha_p[0];
    const float beta  = 1.0f - alpha;

    v8f zero = {};
    v8f acc[8];
    #pragma unroll
    for (int i = 0; i < 8; ++i) acc[i] = zero;
    float rsum = 0.0f;

    const float* Arow = A + (size_t)arow * NDIM;
    const int koff1 = hi ? 16 : 0;
    const int koff2 = koff1 + 8;

    // prologue: stage x-tile 0 and preload A chunk for k=0
    stage_tile(xw, &ldsX[0][0], tid);
    AChunk cur = load_a(Arow, 0, koff1, koff2);

    int p = 0;
    for (int k = 0; k < NDIM; k += 32) {
        const bool hasNext = (k + 32) < NDIM;

        // issue next x-tile copy into the other LDS buffer (async path overlaps)
        if (hasNext)
            stage_tile(xw + (size_t)((k + 32) >> 4) * (8 * 16 * 16), &ldsX[p ^ 1][0], tid);
        if (hasNext) wait_stage_keep4(); else wait_stage_zero();
        __syncthreads();

        // pipeline: next A chunk in flight while we convert/compute current
        AChunk nxt = cur;
        if (hasNext) nxt = load_a(Arow, k + 32, koff1, koff2);
        if (k + 64 < NDIM) {
            __builtin_prefetch(Arow + k + 64, 0, 3);        // global_prefetch, 2 tiles ahead
            __builtin_prefetch(Arow + k + 64 + 16, 0, 3);
        }

        // A fragment: exact fp32 rowsum + bf16 convert
        BFrag a;
        {
            float f[16];
            #pragma unroll
            for (int e = 0; e < 4; ++e) {
                f[e] = cur.v0[e]; f[4+e] = cur.v1[e]; f[8+e] = cur.v2[e]; f[12+e] = cur.v3[e];
            }
            #pragma unroll
            for (int e = 0; e < 16; ++e) { rsum += f[e]; a.h[e] = f2bf(f[e]); }
        }

        // 8 col tiles: B frag = 32 contiguous bytes from LDS, then WMMA
        const unsigned short* xb = &ldsX[p][0];
        #pragma unroll
        for (int ct = 0; ct < 8; ++ct) {
            BFrag b;
            const u32x4* ls = (const u32x4*)(xb + (size_t)((hi * 8 + ct) * 16 + ln) * 16);
            b.q[0] = ls[0];
            b.q[1] = ls[1];
            acc[ct] = __builtin_amdgcn_wmma_f32_16x16x32_bf16(
                false, a.v, false, b.v, (short)0, acc[ct], false, false);
        }
        __syncthreads();

        cur = nxt;
        p ^= 1;
    }

    // ---- rowsum reduce (lane l + lane l^16 cover the full row), row scale
    float rsum_tot = rsum + __shfl_xor(rsum, 16, 32);
    float s   = alpha * rsum_tot + beta;          // rowsum of A' for row rowBase+ln
    float inv = (s == 0.0f) ? 0.0f : 1.0f / s;    // matches inf->0 zeroing

    // ---- x1 = inv * (alpha*y + (1-alpha)*x), write to per-wave LDS tile
    #pragma unroll
    for (int v = 0; v < 8; ++v) {
        int r = v + 8 * hi;                       // C-layout row of VGPR v
        float invv = __shfl(inv, r, 32);
        #pragma unroll
        for (int ct = 0; ct < 8; ++ct) {
            float xo = x[(size_t)(rowBase + r) * HDIM + ct * 16 + ln];
            float y  = alpha * acc[ct][v] + beta * xo;
            ldsY[wave][r][ct * 16 + ln] = invv * y;
        }
    }
    __syncthreads();

    // ---- second GEMM: out = relu(x1 @ W^T + b), K=128 -> 4 steps x 8 tiles
    v8f d[8];
    #pragma unroll
    for (int i = 0; i < 8; ++i) d[i] = zero;

    #pragma unroll
    for (int kt = 0; kt < 4; ++kt) {
        BFrag a2;
        {
            const float* yrow = &ldsY[wave][ln][0];
            const f32x4* q1 = (const f32x4*)(yrow + kt * 32 + koff1);
            const f32x4* q2 = (const f32x4*)(yrow + kt * 32 + koff2);
            f32x4 v0 = q1[0], v1 = q1[1], v2 = q2[0], v3 = q2[1];
            #pragma unroll
            for (int e = 0; e < 4; ++e) {
                a2.h[e]      = f2bf(v0[e]);  a2.h[4 + e]  = f2bf(v1[e]);
                a2.h[8 + e]  = f2bf(v2[e]);  a2.h[12 + e] = f2bf(v3[e]);
            }
        }
        int kb = kt * 2 + hi;
        #pragma unroll
        for (int ct = 0; ct < 8; ++ct) {
            BFrag b2;
            const u32x4* ws_ = (const u32x4*)(ww + (size_t)((kb * 8 + ct) * 16 + ln) * 16);
            b2.q[0] = ws_[0];
            b2.q[1] = ws_[1];
            d[ct] = __builtin_amdgcn_wmma_f32_16x16x32_bf16(
                false, a2.v, false, b2.v, (short)0, d[ct], false, false);
        }
    }

    // ---- bias + relu + store (C layout: VGPR v -> row v+8*hi, col = ct*16+ln)
    #pragma unroll
    for (int ct = 0; ct < 8; ++ct) {
        float bv = bias[ct * 16 + ln];
        #pragma unroll
        for (int v = 0; v < 8; ++v) {
            int r = rowBase + v + 8 * hi;
            float val = d[ct][v] + bv;
            val = val > 0.0f ? val : 0.0f;
            out[(size_t)r * HDIM + ct * 16 + ln] = val;
        }
    }
}

// ---------------------------------------------------------------------------
extern "C" void kernel_launch(void* const* d_in, const int* in_sizes, int n_in,
                              void* d_out, int out_size, void* d_ws, size_t ws_size,
                              hipStream_t stream) {
    // inputs: t, x, A, alpha, W, b  (all fp32)
    const float* x     = (const float*)d_in[1];
    const float* A     = (const float*)d_in[2];
    const float* alpha = (const float*)d_in[3];
    const float* W     = (const float*)d_in[4];
    const float* b     = (const float*)d_in[5];
    float* out = (float*)d_out;

    // workspace: xw bf16 tiles (2 MB) | ww bf16 tiles (32 KB)
    unsigned short* xw = (unsigned short*)d_ws;
    unsigned short* ww = (unsigned short*)((char*)d_ws + (size_t)(2u << 20));

    const int XCHUNKS = (NDIM / 16) * (HDIM / 16) * 16;   // 65536
    const int WCHUNKS = (HDIM / 16) * (HDIM / 16) * 16;   // 1024
    int prep_blocks = (XCHUNKS + WCHUNKS + 255) / 256;    // 260
    prep_kernel<<<prep_blocks, 256, 0, stream>>>(x, W, xw, ww);

    int main_blocks = NDIM / (16 * WAVES);                // 128
    odefunc_kernel<<<main_blocks, 32 * WAVES, 0, stream>>>(A, x, xw, ww, alpha, b, out);
}
